// TrulyNeuralRegisterFile_18975165514096
// MI455X (gfx1250) — matrix operand order repaired and checked by reading, not compiled, not run on hardware
//
#include <hip/hip_runtime.h>
#include <math.h>

#define BATCH   262144
#define BW      64
#define HID     128
#define NREG    32
#define KEYD    256

typedef __attribute__((ext_vector_type(8)))  float          v8f;
typedef __attribute__((ext_vector_type(16))) __bf16         v16bf;
typedef __attribute__((ext_vector_type(16))) unsigned short v16u;
typedef __attribute__((ext_vector_type(8)))  unsigned short v8u;

__device__ __forceinline__ unsigned short f2bf(float f) {
    unsigned int u = __builtin_bit_cast(unsigned int, f);
    return (unsigned short)((u + 0x7FFFu + ((u >> 16) & 1u)) >> 16);  // RNE
}
__device__ __forceinline__ float bf2f(unsigned short h) {
    unsigned int u = ((unsigned int)h) << 16;
    return __builtin_bit_cast(float, u);
}
__device__ __forceinline__ float gelu_exact(float v) {
    return 0.5f * v * (1.0f + erff(v * 0.70710678118654752f));
}

// ---- workspace layout (floats unless noted) ----
#define WS_ATT   0        // 32*32 att_table[idx][reg]
#define WS_CNT   1024     // 32 counts (float)
#define WS_SSUM  1056     // 32*64 enc sums per write idx
#define WS_RTAB  3104     // 32*64 read lookup table
#define WS_SCAL  5152     // [0]=sigmoid(write_strength)
#define WS_W1P_B 20640    // byte offset: 8192 bf16 packed W1 fragments
#define WS_W2P_B 37024    // byte offset: 8192 bf16 packed W2 fragments

// =====================================================================
// Kernel A: att_table for the 32 indices, weight packing (B-fragment
// layout), zero global accumulators, scalars.  1 workgroup.
// =====================================================================
__global__ __launch_bounds__(256) void tnrf_table_kernel(
    const float* __restrict__ qw1, const float* __restrict__ qb1,
    const float* __restrict__ qg1, const float* __restrict__ qbt1,
    const float* __restrict__ qw2, const float* __restrict__ qb2,
    const float* __restrict__ qg2, const float* __restrict__ qbt2,
    const float* __restrict__ qw3, const float* __restrict__ qb3,
    const float* __restrict__ rkeys, const float* __restrict__ temp,
    const float* __restrict__ wstr,
    const float* __restrict__ vw1, const float* __restrict__ vw2,
    float* __restrict__ att, float* __restrict__ gS, float* __restrict__ gCnt,
    float* __restrict__ scal,
    unsigned short* __restrict__ w1p, unsigned short* __restrict__ w2p)
{
    __shared__ float A[NREG * 256];   // pre1/h1 (128 wide) then q/qhat (256 wide)
    __shared__ float Bb[NREG * 256];  // pre2/h2 then key norms in Bb[0..32)
    const int t = threadIdx.x;

    // ---- phase 0 (all threads): zero accumulators, pack weights, scalars
    for (int i = t; i < NREG * BW; i += 256) gS[i] = 0.f;
    if (t < NREG) gCnt[t] = 0.f;
    // W1 pack: [kt(2)][nt(8)][lane(32)][e(16)] ; N=lane&15 ; h=lane>>4: K=8h+e (e<8), K=16+8h+(e-8) (e>=8)
    for (int i = t; i < 2 * 8 * 32 * 16; i += 256) {
        int e = i & 15, ln = (i >> 4) & 31, nt = (i >> 9) & 7, kt = i >> 12;
        int n = nt * 16 + (ln & 15);
        int k = kt * 32 + ((ln >> 4) << 3) + (e & 7) + ((e >> 3) << 4);
        w1p[i] = f2bf(vw1[k * HID + n]);
    }
    // W2 pack: [kt(4)][nt(4)][lane(32)][e(16)]
    for (int i = t; i < 4 * 4 * 32 * 16; i += 256) {
        int e = i & 15, ln = (i >> 4) & 31, nt = (i >> 9) & 3, kt = i >> 11;
        int n = nt * 16 + (ln & 15);
        int k = kt * 32 + ((ln >> 4) << 3) + (e & 7) + ((e >> 3) << 4);
        w2p[i] = f2bf(vw2[k * BW + n]);
    }
    if (t == 0) scal[0] = 1.0f / (1.0f + expf(-wstr[0]));
    __syncthreads();

    // ---- phase 1 (32 threads, one per register index): layer 1 (5 -> 128)
    if (t < NREG) {
        const int i = t;
        float bits[5];
        for (int j = 0; j < 5; ++j) bits[j] = (float)((i >> j) & 1);
        float s1 = 0.f, s2 = 0.f;
        for (int j = 0; j < 128; ++j) {
            float p = qb1[j];
            for (int k = 0; k < 5; ++k) p += bits[k] * qw1[k * 128 + j];
            A[i * 256 + j] = p; s1 += p; s2 += p * p;
        }
        float mu = s1 * (1.f / 128.f);
        float rs = rsqrtf(s2 * (1.f / 128.f) - mu * mu + 1e-5f);
        for (int j = 0; j < 128; ++j) {
            float v = (A[i * 256 + j] - mu) * rs * qg1[j] + qbt1[j];
            A[i * 256 + j] = gelu_exact(v);
        }
    }
    __syncthreads();

    // ---- phase 2 (256 threads, 8 per row): pre2 = h1 @ qw2 (128 -> 256)
    {
        const int i = t >> 3, c = t & 7;
        float s1 = 0.f, s2 = 0.f;
        for (int j = c * 32; j < c * 32 + 32; ++j) {
            float p = qb2[j];
            for (int k = 0; k < 128; ++k) p += A[i * 256 + k] * qw2[k * 256 + j];
            Bb[i * 256 + j] = p; s1 += p; s2 += p * p;
        }
        A[i * 256 + 128 + c] = s1;   // partials live in dead half of A row
        A[i * 256 + 136 + c] = s2;
    }
    __syncthreads();
    {   // LN + GELU on pre2 (redundant stat combine per group of 8)
        const int i = t >> 3, c = t & 7;
        float s1 = 0.f, s2 = 0.f;
        for (int cc = 0; cc < 8; ++cc) { s1 += A[i * 256 + 128 + cc]; s2 += A[i * 256 + 136 + cc]; }
        float mu = s1 * (1.f / 256.f);
        float rs = rsqrtf(s2 * (1.f / 256.f) - mu * mu + 1e-5f);
        for (int j = c * 32; j < c * 32 + 32; ++j) {
            float v = (Bb[i * 256 + j] - mu) * rs * qg2[j] + qbt2[j];
            Bb[i * 256 + j] = gelu_exact(v);
        }
    }
    __syncthreads();

    // ---- phase 3: q = h2 @ qw3 + qb3 (256 -> 256), into A
    {
        const int i = t >> 3, c = t & 7;
        for (int j = c * 32; j < c * 32 + 32; ++j) {
            float p = qb3[j];
            for (int k = 0; k < 256; ++k) p += Bb[i * 256 + k] * qw3[k * 256 + j];
            A[i * 256 + j] = p;
        }
    }
    __syncthreads();
    {   // q norm partials into Bb (h2 dead now)
        const int i = t >> 3, c = t & 7;
        float sq = 0.f;
        for (int j = c * 32; j < c * 32 + 32; ++j) { float v = A[i * 256 + j]; sq += v * v; }
        Bb[i * 256 + c] = sq;
    }
    __syncthreads();
    {   // normalize q
        const int i = t >> 3, c = t & 7;
        float sq = 0.f;
        for (int cc = 0; cc < 8; ++cc) sq += Bb[i * 256 + cc];
        float nq = sqrtf(sq); if (nq < 1e-12f) nq = 1e-12f;
        float rn = 1.f / nq;
        for (int j = c * 32; j < c * 32 + 32; ++j) A[i * 256 + j] *= rn;
    }
    __syncthreads();

    // ---- phase 4: key norms into Bb[0..32)
    if (t < NREG) {
        float sq = 0.f;
        for (int j = 0; j < KEYD; ++j) { float v = rkeys[t * KEYD + j]; sq += v * v; }
        float nk = sqrtf(sq); if (nk < 1e-12f) nk = 1e-12f;
        Bb[t] = nk;
    }
    __syncthreads();

    // ---- phase 5: sim + softmax -> att_table
    if (t < NREG) {
        const int i = t;
        float tp = fabsf(temp[0]); if (tp < 0.01f) tp = 0.01f;
        float invt = 1.f / tp;
        float sim[NREG], mx = -1e30f;
        for (int n = 0; n < NREG; ++n) {
            float d = 0.f;
            for (int k = 0; k < KEYD; ++k) d += A[i * 256 + k] * rkeys[n * KEYD + k];
            d = d / Bb[n] * invt;
            sim[n] = d; mx = fmaxf(mx, d);
        }
        float se = 0.f;
        for (int n = 0; n < NREG; ++n) { sim[n] = expf(sim[n] - mx); se += sim[n]; }
        float rse = 1.f / se;
        for (int n = 0; n < NREG; ++n) att[i * NREG + n] = sim[n] * rse;
    }
}

// =====================================================================
// Kernel B: per-128-row tile: X -> (bf16 WMMA GEMM1) -> LN+GELU ->
// (bf16 WMMA GEMM2) -> scatter-sum by write_idx into global S + counts.
// grid = 2048, block = 256 (8 waves, 16 rows each).
// =====================================================================
__global__ __launch_bounds__(256) void tnrf_enc_kernel(
    const float* __restrict__ wval, const int* __restrict__ widx,
    const float* __restrict__ vb1, const float* __restrict__ vg1,
    const float* __restrict__ vbt1, const float* __restrict__ vb2,
    const unsigned short* __restrict__ w1p, const unsigned short* __restrict__ w2p,
    float* __restrict__ gS, float* __restrict__ gCnt)
{
    __shared__ __align__(16) unsigned short Xs[128 * BW];    // bf16 X tile
    __shared__ __align__(16) unsigned short Hs[128 * HID];   // bf16 hidden tile
    __shared__ float Ssum[NREG * BW];
    __shared__ float Cnt[NREG];
    __shared__ int   Wi[128];

    const int t = threadIdx.x;
    const int lane = t & 31;
    const int w = t >> 5;                       // wave id: rows [16w,16w+16)
    const long row0 = (long)blockIdx.x * 128;

    for (int i = t; i < NREG * BW; i += 256) Ssum[i] = 0.f;
    if (t < NREG) Cnt[t] = 0.f;
    if (t < 128) Wi[t] = widx[row0 + t];
    __syncthreads();

    // ---- load X (mask idx==31), convert to bf16
    {
        const int r = t >> 1, hh = (t & 1) * 32;
        const float m = (Wi[r] == 31) ? 0.f : 1.f;
        const float4* s4 = (const float4*)(wval + (row0 + r) * BW + hh);
        for (int j = 0; j < 8; ++j) {
            float4 v = s4[j];
            const int b = r * BW + hh + j * 4;
            Xs[b + 0] = f2bf(v.x * m); Xs[b + 1] = f2bf(v.y * m);
            Xs[b + 2] = f2bf(v.z * m); Xs[b + 3] = f2bf(v.w * m);
        }
    }
    __syncthreads();

    // ---- GEMM1: X(16x64) @ W1(64x128), per wave
    {
        v8f acc[8];
        for (int nt = 0; nt < 8; ++nt)
            for (int e = 0; e < 8; ++e) acc[nt][e] = 0.f;
        for (int kt = 0; kt < 2; ++kt) {
            const unsigned short* ap =
                &Xs[(w * 16 + (lane & 15)) * BW + kt * 32 + ((lane >> 4) << 3)];
            v8u lo = *(const v8u*)ap;
            v8u hi = *(const v8u*)(ap + 16);
            v16u av = __builtin_shufflevector(lo, hi, 0,1,2,3,4,5,6,7,8,9,10,11,12,13,14,15);
            v16bf a = __builtin_bit_cast(v16bf, av);
            for (int nt = 0; nt < 8; ++nt) {
                v16u bv = *(const v16u*)&w1p[((kt * 8 + nt) * 32 + lane) * 16];
                v16bf b = __builtin_bit_cast(v16bf, bv);
                acc[nt] = __builtin_amdgcn_wmma_f32_16x16x32_bf16(
                    false, a, false, b, (short)0, acc[nt], false, false);
            }
        }
        for (int nt = 0; nt < 8; ++nt)
            for (int r = 0; r < 8; ++r) {
                const int rr = w * 16 + ((lane < 16) ? r : r + 8);
                const int cc = nt * 16 + (lane & 15);
                Hs[rr * HID + cc] = f2bf(acc[nt][r]);
            }
    }
    __syncthreads();

    // ---- bias + LayerNorm + exact GELU, one thread per row
    if (t < 128) {
        float s1 = 0.f, s2 = 0.f;
        for (int j = 0; j < HID; ++j) {
            float h = bf2f(Hs[t * HID + j]) + vb1[j];
            s1 += h; s2 += h * h;
        }
        float mu = s1 * (1.f / HID);
        float rs = rsqrtf(s2 * (1.f / HID) - mu * mu + 1e-5f);
        for (int j = 0; j < HID; ++j) {
            float h = bf2f(Hs[t * HID + j]) + vb1[j];
            float v = (h - mu) * rs * vg1[j] + vbt1[j];
            Hs[t * HID + j] = f2bf(gelu_exact(v));
        }
    }
    __syncthreads();

    // ---- GEMM2: H(16x128) @ W2(128x64) + scatter into Ssum
    {
        v8f acc[4];
        for (int nt = 0; nt < 4; ++nt)
            for (int e = 0; e < 8; ++e) acc[nt][e] = 0.f;
        for (int kt = 0; kt < 4; ++kt) {
            const unsigned short* ap =
                &Hs[(w * 16 + (lane & 15)) * HID + kt * 32 + ((lane >> 4) << 3)];
            v8u lo = *(const v8u*)ap;
            v8u hi = *(const v8u*)(ap + 16);
            v16u av = __builtin_shufflevector(lo, hi, 0,1,2,3,4,5,6,7,8,9,10,11,12,13,14,15);
            v16bf a = __builtin_bit_cast(v16bf, av);
            for (int nt = 0; nt < 4; ++nt) {
                v16u bv = *(const v16u*)&w2p[((kt * 4 + nt) * 32 + lane) * 16];
                v16bf b = __builtin_bit_cast(v16bf, bv);
                acc[nt] = __builtin_amdgcn_wmma_f32_16x16x32_bf16(
                    false, a, false, b, (short)0, acc[nt], false, false);
            }
        }
        for (int nt = 0; nt < 4; ++nt)
            for (int r = 0; r < 8; ++r) {
                const int rr = w * 16 + ((lane < 16) ? r : r + 8);
                const int cc = nt * 16 + (lane & 15);
                float val = acc[nt][r] + vb2[cc];
                atomicAdd(&Ssum[Wi[rr] * BW + cc], val);
            }
    }
    if (t < 128) atomicAdd(&Cnt[Wi[t]], 1.0f);
    __syncthreads();

    for (int i = t; i < NREG * BW; i += 256) atomicAdd(&gS[i], Ssum[i]);
    if (t < NREG) atomicAdd(&gCnt[t], Cnt[t]);
}

// =====================================================================
// Kernel C: new_regs (32x64) + read lookup R_table (32x64). 1 workgroup.
// =====================================================================
__global__ __launch_bounds__(256) void tnrf_combine_kernel(
    const float* __restrict__ att, const float* __restrict__ gS,
    const float* __restrict__ gCnt, const float* __restrict__ scal,
    const float* __restrict__ regvals,
    float* __restrict__ out_newregs, float* __restrict__ rtab)
{
    __shared__ float nr[NREG * BW];
    __shared__ float matt[NREG];
    const int t = threadIdx.x;
    const float s = scal[0];
    const float invB = 1.0f / (float)BATCH;

    if (t < NREG) {
        float a = 0.f;
        for (int i = 0; i < NREG; ++i) a += gCnt[i] * att[i * NREG + t];
        matt[t] = a * invB;
    }
    __syncthreads();
    for (int e = t; e < NREG * BW; e += 256) {
        const int n = e >> 6, d = e & 63;
        float acc = 0.f;
        for (int i = 0; i < NREG; ++i) acc += att[i * NREG + n] * gS[i * BW + d];
        float v = (1.f - s * matt[n]) * regvals[e] + (s * invB) * acc;
        nr[e] = v;
        out_newregs[e] = v;
    }
    __syncthreads();
    for (int e = t; e < NREG * BW; e += 256) {
        const int i = e >> 6, d = e & 63;
        float r = 0.f;
        if (i != 31)
            for (int n = 0; n < NREG; ++n) r += att[i * NREG + n] * nr[n * BW + d];
        rtab[e] = r;
    }
}

// =====================================================================
// Kernel D: read gather — pure bandwidth. grid = 1024, block = 256.
// =====================================================================
__global__ __launch_bounds__(256) void tnrf_read_kernel(
    const float* __restrict__ rtab, const int* __restrict__ ridx,
    float* __restrict__ out)
{
    __shared__ __align__(16) float Rt[NREG * BW];
    const int t = threadIdx.x;
    for (int e = t; e < NREG * BW; e += 256) Rt[e] = rtab[e];
    __syncthreads();
    const long row = (long)blockIdx.x * 256 + t;
    const int idx = ridx[row];
    const float4* src = (const float4*)&Rt[idx * BW];
    float4* dst = (float4*)(out + row * BW);
    for (int j = 0; j < 16; ++j) dst[j] = src[j];
}

// =====================================================================
extern "C" void kernel_launch(void* const* d_in, const int* in_sizes, int n_in,
                              void* d_out, int out_size, void* d_ws, size_t ws_size,
                              hipStream_t stream) {
    const int*   write_idx       = (const int*)  d_in[0];
    const float* write_value     = (const float*)d_in[1];
    const int*   read_idx        = (const int*)  d_in[2];
    const float* register_values = (const float*)d_in[3];
    const float* register_keys   = (const float*)d_in[4];
    const float* qw1  = (const float*)d_in[5];
    const float* qb1  = (const float*)d_in[6];
    const float* qg1  = (const float*)d_in[7];
    const float* qbt1 = (const float*)d_in[8];
    const float* qw2  = (const float*)d_in[9];
    const float* qb2  = (const float*)d_in[10];
    const float* qg2  = (const float*)d_in[11];
    const float* qbt2 = (const float*)d_in[12];
    const float* qw3  = (const float*)d_in[13];
    const float* qb3  = (const float*)d_in[14];
    const float* vw1  = (const float*)d_in[15];
    const float* vb1  = (const float*)d_in[16];
    const float* vg1  = (const float*)d_in[17];
    const float* vbt1 = (const float*)d_in[18];
    const float* vw2  = (const float*)d_in[19];
    const float* vb2  = (const float*)d_in[20];
    const float* temperature    = (const float*)d_in[21];
    const float* write_strength = (const float*)d_in[22];

    float* ws   = (float*)d_ws;
    float* att  = ws + WS_ATT;
    float* gCnt = ws + WS_CNT;
    float* gS   = ws + WS_SSUM;
    float* rtab = ws + WS_RTAB;
    float* scal = ws + WS_SCAL;
    unsigned short* w1p = (unsigned short*)((char*)d_ws + WS_W1P_B);
    unsigned short* w2p = (unsigned short*)((char*)d_ws + WS_W2P_B);

    float* out_read = (float*)d_out;                          // (B, 64)
    float* out_regs = out_read + (size_t)BATCH * BW;          // (32, 64)

    tnrf_table_kernel<<<1, 256, 0, stream>>>(
        qw1, qb1, qg1, qbt1, qw2, qb2, qg2, qbt2, qw3, qb3,
        register_keys, temperature, write_strength, vw1, vw2,
        att, gS, gCnt, scal, w1p, w2p);

    tnrf_enc_kernel<<<BATCH / 128, 256, 0, stream>>>(
        write_value, write_idx, vb1, vg1, vbt1, vb2, w1p, w2p, gS, gCnt);

    tnrf_combine_kernel<<<1, 256, 0, stream>>>(
        att, gS, gCnt, scal, register_values, out_regs, rtab);

    tnrf_read_kernel<<<BATCH / 256, 256, 0, stream>>>(rtab, read_idx, out_read);
}
